// MultiHeadAttention_65652870087112
// MI455X (gfx1250) — compile-verified, matrix operand order checked by reading, not run on hardware
//
#include <hip/hip_runtime.h>
#include <hip/hip_bf16.h>

typedef __bf16 bf16;
typedef __attribute__((ext_vector_type(16))) __bf16 v16bf;
typedef __attribute__((ext_vector_type(8)))  __bf16 v8bf;
typedef __attribute__((ext_vector_type(4)))  __bf16 v4bf;
typedef __attribute__((ext_vector_type(8)))  float   v8f;
typedef __attribute__((ext_vector_type(4)))  float   v4f;
typedef __attribute__((ext_vector_type(4)))  int     v4i;

union FragBF { v16bf v; v8bf h[2]; };

#define WMMA_BF16(a, b, c) \
  __builtin_amdgcn_wmma_f32_16x16x32_bf16(false, (a), false, (b), (short)0, (c), false, false)

// ---- async global->LDS copy (ASYNCcnt path), with safe fallback -------------
#if defined(__gfx1250__) && __has_builtin(__builtin_amdgcn_global_load_async_to_lds_b128)
typedef __attribute__((address_space(1))) v4i as1_v4i;
typedef __attribute__((address_space(3))) v4i as3_v4i;
static __device__ __forceinline__ void cp_async16(void* lds, const void* glb) {
  __builtin_amdgcn_global_load_async_to_lds_b128(
      (as1_v4i*)(glb), (as3_v4i*)(lds), 0, 0);
}
#else
static __device__ __forceinline__ void cp_async16(void* lds, const void* glb) {
  *(v8bf*)lds = *(const v8bf*)glb;
}
#endif

static __device__ __forceinline__ void wait_async0() {
#if defined(__gfx1250__) && __has_builtin(__builtin_amdgcn_s_wait_asynccnt)
  __builtin_amdgcn_s_wait_asynccnt(0);
#else
  asm volatile("s_wait_asynccnt 0" ::: "memory");
#endif
}

// ---------------------------------------------------------------- prep ------
__global__ __launch_bounds__(256) void cvt_f32_bf16(const float* __restrict__ x,
                                                    bf16* __restrict__ y, int n) {
  int i = (blockIdx.x * 256 + threadIdx.x) * 4;
  if (i + 3 < n) {
    v4f v = *(const v4f*)(x + i);
    v4bf o;
    o[0] = (bf16)v[0]; o[1] = (bf16)v[1]; o[2] = (bf16)v[2]; o[3] = (bf16)v[3];
    *(v4bf*)(y + i) = o;
  }
}

// W [K=1024][N=1024] fp32 row-major  ->  Wt [N][K] bf16
__global__ __launch_bounds__(256) void transpose_w_bf16(const float* __restrict__ W,
                                                        bf16* __restrict__ Wt) {
  __shared__ float tile[32][33];
  const int j  = threadIdx.x & 31;
  const int i0 = (threadIdx.x >> 5) * 4;
  const int kb = blockIdx.x * 32, nb = blockIdx.y * 32;
#pragma unroll
  for (int ii = 0; ii < 4; ++ii)
    tile[i0 + ii][j] = W[(size_t)(kb + i0 + ii) * 1024 + nb + j];
  __syncthreads();
#pragma unroll
  for (int ii = 0; ii < 4; ++ii)
    Wt[(size_t)(nb + i0 + ii) * 1024 + kb + j] = (bf16)tile[j][i0 + ii];
}

// ---------------------------------------------------------------- GEMM ------
// C[M=4096][N=1024] = A[M][K=1024] * Bt[N][K]^T + bias
// Block 128x128, 8 waves, wave tile 32x64, K-step 32, double-buffered async LDS.
// mode 0: bf16 store to [B,H,S,DK]   (Q, K)
// mode 1: bf16 store to [B,H,DK,S]   (V transposed)
// mode 2: fp32 store to [M][N]       (final output)
#define LDA 40   // padded LDS row stride (bf16 elems): 80B, keeps 16B alignment

__global__ __launch_bounds__(256) void gemm_bf16(const bf16* __restrict__ A,
                                                 const bf16* __restrict__ Bt,
                                                 const float* __restrict__ bias,
                                                 void* __restrict__ outp, int mode) {
  __shared__ bf16 As[2][128 * LDA];
  __shared__ bf16 Bs[2][128 * LDA];
  const int tid  = threadIdx.x;
  const int lane = tid & 31;
  const int wid  = tid >> 5;          // 0..7
  const int wm   = (wid >> 1) * 32;   // 0,32,64,96
  const int wn   = (wid & 1) * 64;    // 0,64
  const int r    = lane & 15;
  const int hi   = lane >> 4;
  const int m0   = blockIdx.x * 128;
  const int n0   = blockIdx.y * 128;

  const int srow = tid >> 1, scol = (tid & 1) * 16;   // 128 rows x 32 k, 32B/thread

  v8f acc[2][4];
#pragma unroll
  for (int i = 0; i < 2; ++i)
#pragma unroll
    for (int j = 0; j < 4; ++j)
#pragma unroll
      for (int e = 0; e < 8; ++e) acc[i][j][e] = 0.0f;

  const bf16* agBase = A  + (size_t)(m0 + srow) * 1024 + scol;
  const bf16* bgBase = Bt + (size_t)(n0 + srow) * 1024 + scol;
  bf16* asDst0 = &As[0][srow * LDA + scol];
  bf16* asDst1 = &As[1][srow * LDA + scol];
  bf16* bsDst0 = &Bs[0][srow * LDA + scol];
  bf16* bsDst1 = &Bs[1][srow * LDA + scol];

  // stage K-tile `k0` into buffer `buf`
  auto stage = [&](int buf, int k0) {
    bf16* ad = buf ? asDst1 : asDst0;
    bf16* bd = buf ? bsDst1 : bsDst0;
    cp_async16(ad,     agBase + k0);
    cp_async16(ad + 8, agBase + k0 + 8);
    cp_async16(bd,     bgBase + k0);
    cp_async16(bd + 8, bgBase + k0 + 8);
  };

  stage(0, 0);
  wait_async0();
  __syncthreads();

  int cur = 0;
  for (int it = 0; it < 32; ++it) {
    if (it + 1 < 32) stage(cur ^ 1, (it + 1) * 32);   // prefetch next tile

    const bf16* asP = &As[cur][0];
    const bf16* bsP = &Bs[cur][0];
    FragBF af[2], bfr[4];
#pragma unroll
    for (int i = 0; i < 2; ++i) {
      const bf16* p = asP + (wm + i * 16 + r) * LDA + hi * 8;
      af[i].h[0] = *(const v8bf*)p;
      af[i].h[1] = *(const v8bf*)(p + 16);
    }
#pragma unroll
    for (int j = 0; j < 4; ++j) {
      const bf16* p = bsP + (wn + j * 16 + r) * LDA + hi * 8;
      bfr[j].h[0] = *(const v8bf*)p;
      bfr[j].h[1] = *(const v8bf*)(p + 16);
    }
#pragma unroll
    for (int i = 0; i < 2; ++i)
#pragma unroll
      for (int j = 0; j < 4; ++j)
        acc[i][j] = WMMA_BF16(af[i].v, bfr[j].v, acc[i][j]);

    wait_async0();
    __syncthreads();
    cur ^= 1;
  }

#pragma unroll
  for (int i = 0; i < 2; ++i) {
#pragma unroll
    for (int j = 0; j < 4; ++j) {
      const int nG = n0 + wn + j * 16 + r;
      const float bv = bias[nG];
#pragma unroll
      for (int e = 0; e < 8; ++e) {
        const int mG = m0 + wm + i * 16 + hi * 8 + e;
        const float val = acc[i][j][e] + bv;
        if (mode == 2) {
          ((float*)outp)[(size_t)mG * 1024 + nG] = val;
        } else {
          const int bb = mG >> 11, ss = mG & 2047;
          const int hh = nG >> 6,  dk = nG & 63;
          bf16* o = (bf16*)outp;
          if (mode == 0)
            o[(((size_t)(bb * 16 + hh) * 2048) + ss) * 64 + dk] = (bf16)val;
          else
            o[(((size_t)(bb * 16 + hh) * 64) + dk) * 2048 + ss] = (bf16)val;
        }
      }
    }
  }
}

// ----------------------------------------------------------- attention ------
// Q,K: [BH=32][S=2048][DK=64] bf16 ; Vt: [BH][DK=64][S] bf16 ; ctx: [B][S][D] bf16
__global__ __launch_bounds__(128) void attn_kernel(const bf16* __restrict__ Q,
                                                   const bf16* __restrict__ K,
                                                   const bf16* __restrict__ Vt,
                                                   bf16* __restrict__ ctx) {
  __shared__ bf16 Pl[4][16 * LDA];
  const int lane = threadIdx.x & 31;
  const int wid  = threadIdx.x >> 5;   // 0..3
  const int r    = lane & 15;
  const int hi   = lane >> 4;
  const int bh   = blockIdx.y;         // 0..31
  const int b    = bh >> 4;
  const int h    = bh & 15;
  const int q0   = blockIdx.x * 64 + wid * 16;   // this wave's first q row

  const bf16* Qh  = Q  + (size_t)bh * 2048 * 64;
  const bf16* Kh  = K  + (size_t)bh * 2048 * 64;
  const bf16* Vth = Vt + (size_t)bh * 64 * 2048;
  bf16* ctxB = ctx + (size_t)b * 2048 * 1024 + (size_t)h * 64;

  FragBF qf0, qf1;
  {
    const bf16* qp = Qh + (size_t)(q0 + r) * 64 + hi * 8;
    qf0.h[0] = *(const v8bf*)qp;        qf0.h[1] = *(const v8bf*)(qp + 16);
    qf1.h[0] = *(const v8bf*)(qp + 32); qf1.h[1] = *(const v8bf*)(qp + 48);
  }

  v8f o[4];
  float mrow[8], lrow[8];
#pragma unroll
  for (int e = 0; e < 8; ++e) { mrow[e] = -1e30f; lrow[e] = 0.0f; }
#pragma unroll
  for (int d = 0; d < 4; ++d)
#pragma unroll
    for (int e = 0; e < 8; ++e) o[d][e] = 0.0f;

  const int nT = (q0 + 15) / 32 + 1;   // causal: only kv tiles <= diagonal
  for (int kt = 0; kt < nT; ++kt) {
    const int kv0 = kt * 32;
    if (kv0 + 32 < 2048)
      __builtin_prefetch(Kh + (size_t)(kv0 + 32 + r) * 64, 0, 0);

    // ---- scores: S = Q(16xDK) * K^T(DKx32), 4 WMMAs over DK=64, 2 n-tiles
    v8f s[2];
#pragma unroll
    for (int nt = 0; nt < 2; ++nt) {
      const bf16* kp = Kh + (size_t)(kv0 + nt * 16 + r) * 64 + hi * 8;
      FragBF k0f, k1f;
      k0f.h[0] = *(const v8bf*)kp;        k0f.h[1] = *(const v8bf*)(kp + 16);
      k1f.h[0] = *(const v8bf*)(kp + 32); k1f.h[1] = *(const v8bf*)(kp + 48);
      v8f z;
#pragma unroll
      for (int e = 0; e < 8; ++e) z[e] = 0.0f;
      z = WMMA_BF16(qf0.v, k0f.v, z);
      z = WMMA_BF16(qf1.v, k1f.v, z);
      s[nt] = z;
    }

    // ---- scale + causal mask (C layout: lane holds n = r, m = hi*8 + e)
#pragma unroll
    for (int nt = 0; nt < 2; ++nt) {
      const int kvIdx = kv0 + nt * 16 + r;
#pragma unroll
      for (int e = 0; e < 8; ++e) {
        const float sv = s[nt][e] * 0.125f;   // 1/sqrt(DK)
        s[nt][e] = (kvIdx > q0 + hi * 8 + e) ? -1e9f : sv;
      }
    }

    // ---- online softmax per row (reduce across 16-lane group)
#pragma unroll
    for (int e = 0; e < 8; ++e) {
      float v = fmaxf(s[0][e], s[1][e]);
      v = fmaxf(v, __shfl_xor(v, 1, 32));
      v = fmaxf(v, __shfl_xor(v, 2, 32));
      v = fmaxf(v, __shfl_xor(v, 4, 32));
      v = fmaxf(v, __shfl_xor(v, 8, 32));
      const float mnew = fmaxf(mrow[e], v);
      const float corr = __expf(mrow[e] - mnew);
      mrow[e] = mnew;
      const float p0 = __expf(s[0][e] - mnew);
      const float p1 = __expf(s[1][e] - mnew);
      s[0][e] = p0; s[1][e] = p1;
      float ps = p0 + p1;
      ps += __shfl_xor(ps, 1, 32);
      ps += __shfl_xor(ps, 2, 32);
      ps += __shfl_xor(ps, 4, 32);
      ps += __shfl_xor(ps, 8, 32);
      lrow[e] = lrow[e] * corr + ps;
#pragma unroll
      for (int d = 0; d < 4; ++d) o[d][e] *= corr;
    }

    // ---- P: C-layout -> A-layout via per-wave LDS region (no block barrier)
#pragma unroll
    for (int nt = 0; nt < 2; ++nt)
#pragma unroll
      for (int e = 0; e < 8; ++e)
        Pl[wid][(hi * 8 + e) * LDA + nt * 16 + r] = (bf16)s[nt][e];
    asm volatile("s_wait_dscnt 0" ::: "memory");
    FragBF pf;
    {
      const bf16* pp = &Pl[wid][r * LDA + hi * 8];
      pf.h[0] = *(const v8bf*)pp;
      pf.h[1] = *(const v8bf*)(pp + 16);
    }
    asm volatile("s_wait_dscnt 0" ::: "memory");

    // ---- O += P(16x32) * V(32x16) per DK n-tile; Vt rows are contiguous
#pragma unroll
    for (int d = 0; d < 4; ++d) {
      const bf16* vp = Vth + (size_t)(d * 16 + r) * 2048 + kv0 + hi * 8;
      FragBF vf;
      vf.h[0] = *(const v8bf*)vp;
      vf.h[1] = *(const v8bf*)(vp + 16);
      o[d] = WMMA_BF16(pf.v, vf.v, o[d]);
    }
  }

  float inv[8];
#pragma unroll
  for (int e = 0; e < 8; ++e) inv[e] = 1.0f / lrow[e];
#pragma unroll
  for (int d = 0; d < 4; ++d)
#pragma unroll
    for (int e = 0; e < 8; ++e)
      ctxB[(size_t)(q0 + hi * 8 + e) * 1024 + d * 16 + r] = (bf16)(o[d][e] * inv[e]);
}

// ---------------------------------------------------------------- launch ----
extern "C" void kernel_launch(void* const* d_in, const int* in_sizes, int n_in,
                              void* d_out, int out_size, void* d_ws, size_t ws_size,
                              hipStream_t stream) {
  const float* x  = (const float*)d_in[0];
  // d_in[1] = causal mask (tril) -- implemented analytically in attn_kernel
  const float* wq = (const float*)d_in[2];
  const float* bq = (const float*)d_in[3];
  const float* wk = (const float*)d_in[4];
  const float* bk = (const float*)d_in[5];
  const float* wv = (const float*)d_in[6];
  const float* bv = (const float*)d_in[7];
  const float* wo = (const float*)d_in[8];
  const float* bo = (const float*)d_in[9];

  char* ws = (char*)d_ws;
  bf16* xb  = (bf16*)(ws + (0ull  << 20));   // 8 MiB  [4096][1024]
  bf16* wqt = (bf16*)(ws + (8ull  << 20));   // 2 MiB  [N][K]
  bf16* wkt = (bf16*)(ws + (10ull << 20));
  bf16* wvt = (bf16*)(ws + (12ull << 20));
  bf16* wot = (bf16*)(ws + (14ull << 20));
  bf16* Qb  = (bf16*)(ws + (16ull << 20));   // 8 MiB  [BH][S][DK]
  bf16* Kb  = (bf16*)(ws + (24ull << 20));
  bf16* Vtb = (bf16*)(ws + (32ull << 20));   // 8 MiB  [BH][DK][S]
  bf16* ctx = (bf16*)(ws + (40ull << 20));   // 8 MiB  [B*S][D]

  cvt_f32_bf16<<<4096, 256, 0, stream>>>(x, xb, 4096 * 1024);
  dim3 tg(32, 32);
  transpose_w_bf16<<<tg, 256, 0, stream>>>(wq, wqt);
  transpose_w_bf16<<<tg, 256, 0, stream>>>(wk, wkt);
  transpose_w_bf16<<<tg, 256, 0, stream>>>(wv, wvt);
  transpose_w_bf16<<<tg, 256, 0, stream>>>(wo, wot);

  dim3 gg(32, 8);   // M/128 x N/128
  gemm_bf16<<<gg, 256, 0, stream>>>(xb, wqt, bq, (void*)Qb, 0);
  gemm_bf16<<<gg, 256, 0, stream>>>(xb, wkt, bk, (void*)Kb, 0);
  gemm_bf16<<<gg, 256, 0, stream>>>(xb, wvt, bv, (void*)Vtb, 1);

  attn_kernel<<<dim3(32, 32), 128, 0, stream>>>(Qb, Kb, Vtb, ctx);

  gemm_bf16<<<gg, 256, 0, stream>>>(ctx, wot, bo, d_out, 2);
}
